// MultiHeadAttention_56891136803404
// MI455X (gfx1250) — compile-verified, matrix operand order checked by reading, not run on hardware
//
#include <hip/hip_runtime.h>

#define S_LEN  4096
#define D_DIM  1024
#define H_NUM  8
#define DH_DIM 128

typedef unsigned short u16;
typedef __attribute__((ext_vector_type(16))) __bf16 v16bf;
typedef __attribute__((ext_vector_type(8)))  __bf16 v8bf;
typedef __attribute__((ext_vector_type(8)))  float  v8f;
typedef __attribute__((__vector_size__(4 * sizeof(int)))) int v4i;

// gfx1250 async global->LDS path (ASYNCcnt-tracked), with safe fallback.
#if defined(__has_builtin)
#  if __has_builtin(__builtin_amdgcn_global_load_async_to_lds_b128)
#    define HAVE_ASYNC_LDS 1
#  endif
#endif
#ifndef HAVE_ASYNC_LDS
#  define HAVE_ASYNC_LDS 0
#endif

typedef __attribute__((address_space(1))) v4i gv4i;
typedef __attribute__((address_space(3))) v4i lv4i;

__device__ __forceinline__ void async_copy16(const u16* src, u16* lds_dst) {
#if HAVE_ASYNC_LDS
  __builtin_amdgcn_global_load_async_to_lds_b128((gv4i*)src, (lv4i*)lds_dst, 0, 0);
#else
  *reinterpret_cast<uint4*>(lds_dst) = *reinterpret_cast<const uint4*>(src);
#endif
}

__device__ __forceinline__ void async_wait0() {
#if HAVE_ASYNC_LDS
#  if __has_builtin(__builtin_amdgcn_s_wait_asynccnt)
  __builtin_amdgcn_s_wait_asynccnt(0);
#  else
  asm volatile("s_wait_asynccnt 0" ::: "memory");
#  endif
#endif
}

__device__ __forceinline__ u16 f32_to_bf16(float f) {
  unsigned u = __float_as_uint(f);
  u += 0x7fffu + ((u >> 16) & 1u);   // round-to-nearest-even
  return (u16)(u >> 16);
}

// Load one 16x32 bf16 A/B fragment row-slice per the CDNA5 WMMA layout:
// lanes 0-15 hold K = {0..7, 16..23}, lanes 16-31 hold K = {8..15, 24..31}.
__device__ __forceinline__ v16bf frag_ld(const u16* rowp, int half) {
  union { v16bf v; v8bf h[2]; } u;
  u.h[0] = *reinterpret_cast<const v8bf*>(rowp + half * 8);
  u.h[1] = *reinterpret_cast<const v8bf*>(rowp + half * 8 + 16);
  return u.v;
}

__device__ __forceinline__ v8f wmma_bf16(v16bf a, v16bf b, v8f c) {
  return __builtin_amdgcn_wmma_f32_16x16x32_bf16(
      /*neg_a=*/false, a, /*neg_b=*/false, b,
      /*c_mod=*/(short)0, c, /*reuse_a=*/false, /*reuse_b=*/false);
}

// ---------------------------------------------------------------- convert
__global__ __launch_bounds__(256) void cvt_kernel(const float* __restrict__ in,
                                                  u16* __restrict__ out, int n) {
  int i = blockIdx.x * 256 + threadIdx.x;
  if (i < n) out[i] = f32_to_bf16(in[i]);
}

// ---------------------------------------------------------------- GEMM core
// 128x128 macro tile, 256 threads (8 waves), double-buffered LDS.
// a_lds/bt_lds: [2*128][40] (+8 pad => conflict-free 16B fragment reads).
__device__ __forceinline__ void gemm_accum(
    const u16* __restrict__ A, int lda,
    const u16* __restrict__ B, int ldb,
    int m0, int n0, int K,
    u16 (*a_lds)[40], u16 (*bt_lds)[40],
    v8f acc[4][2]) {
  const int t = threadIdx.x;
  const int w = t >> 5, lane = t & 31, lo = lane & 15, half = lane >> 4;
  const int wm = w >> 2, wn = w & 3;

  auto stage = [&](int k0, int buf) {
    // A: 128x32 row-major, two 16B async copies per thread
    int row = t >> 1, colb = (t & 1) * 16;
    const u16* ag = A + (size_t)(m0 + row) * lda + (k0 + colb);
    u16* ad = &a_lds[buf * 128 + row][colb];
    async_copy16(ag, ad);
    async_copy16(ag + 8, ad + 8);
    // B: 32x128 -> transposed bt[col][k] (needs registers for transpose)
    int kr = t >> 3, cb = (t & 7) * 16;
    const u16* bg = B + (size_t)(k0 + kr) * ldb + (n0 + cb);
    uint4 p0 = *reinterpret_cast<const uint4*>(bg);
    uint4 p1 = *reinterpret_cast<const uint4*>(bg + 8);
    const u16* q0 = reinterpret_cast<const u16*>(&p0);
    const u16* q1 = reinterpret_cast<const u16*>(&p1);
#pragma unroll
    for (int j = 0; j < 8; ++j) {
      bt_lds[buf * 128 + cb + j][kr]     = q0[j];
      bt_lds[buf * 128 + cb + 8 + j][kr] = q1[j];
    }
  };

  stage(0, 0);
  for (int k0 = 0; k0 < K; k0 += 32) {
    const int cur = (k0 >> 5) & 1;
    async_wait0();           // own async copies landed in LDS
    __syncthreads();         // everyone's tile (incl. ds_store Bt) visible
    if (k0 + 32 < K) stage(k0 + 32, cur ^ 1);  // prefetch next into other buf

    v16bf af[4], bf[2];
#pragma unroll
    for (int mi = 0; mi < 4; ++mi)
      af[mi] = frag_ld(&a_lds[cur * 128 + wm * 64 + mi * 16 + lo][0], half);
#pragma unroll
    for (int ni = 0; ni < 2; ++ni)
      bf[ni] = frag_ld(&bt_lds[cur * 128 + wn * 32 + ni * 16 + lo][0], half);
#pragma unroll
    for (int mi = 0; mi < 4; ++mi)
#pragma unroll
      for (int ni = 0; ni < 2; ++ni)
        acc[mi][ni] = wmma_bf16(af[mi], bf[ni], acc[mi][ni]);
  }
}

// ---------------------------------------------------------------- QKV proj
__global__ __launch_bounds__(256) void qkv_proj_kernel(
    const u16* __restrict__ z16,
    const u16* __restrict__ wq, const u16* __restrict__ wk, const u16* __restrict__ wv,
    const float* __restrict__ bq, const float* __restrict__ bk, const float* __restrict__ bv,
    u16* __restrict__ qo, u16* __restrict__ ko, u16* __restrict__ vo) {
  __shared__ u16 a_lds[256][40];
  __shared__ u16 bt_lds[256][40];
  const int sel = blockIdx.z, h = blockIdx.y, m0 = blockIdx.x * 128;
  const u16* W = (sel == 0 ? wq : sel == 1 ? wk : wv) + (size_t)h * D_DIM * DH_DIM;
  const float* bias = (sel == 0 ? bq : sel == 1 ? bk : bv) + (size_t)h * DH_DIM;
  u16* out = (sel == 0 ? qo : sel == 1 ? ko : vo) + (size_t)h * S_LEN * DH_DIM;
  const float scale = (sel == 0) ? 0.08838834764831845f : 1.0f;  // 1/sqrt(128) folded into Q

  v8f acc[4][2] = {};
  gemm_accum(z16, D_DIM, W, DH_DIM, m0, 0, D_DIM, a_lds, bt_lds, acc);

  const int t = threadIdx.x, w = t >> 5, lane = t & 31, lo = lane & 15, half = lane >> 4;
  const int wm = w >> 2, wn = w & 3;
#pragma unroll
  for (int mi = 0; mi < 4; ++mi)
#pragma unroll
    for (int ni = 0; ni < 2; ++ni)
#pragma unroll
      for (int r = 0; r < 8; ++r) {
        int grow = m0 + wm * 64 + mi * 16 + half * 8 + r;
        int gcol = wn * 32 + ni * 16 + lo;
        float val = (acc[mi][ni][r] + bias[gcol]) * scale;
        out[(size_t)grow * DH_DIM + gcol] = f32_to_bf16(val);
      }
}

// ---------------------------------------------------------------- attention
__global__ __launch_bounds__(256) void attn_kernel(
    const u16* __restrict__ q16, const u16* __restrict__ k16,
    const u16* __restrict__ v16, u16* __restrict__ cat16) {
  __shared__ u16 kt[64][136];     // K block [key][dh]      (+8 pad)
  __shared__ u16 vt[128][72];     // V block transposed [dh][key]
  __shared__ u16 pl[8][16][72];   // per-wave P scratch [qrow][key]

  const int h = blockIdx.y, qblk = blockIdx.x;
  const u16* qh = q16 + (size_t)h * S_LEN * DH_DIM;
  const u16* kh = k16 + (size_t)h * S_LEN * DH_DIM;
  const u16* vh = v16 + (size_t)h * S_LEN * DH_DIM;

  const int t = threadIdx.x, w = t >> 5, lane = t & 31, lo = lane & 15, half = lane >> 4;
  const int qrow = qblk * 128 + w * 16 + lo;

  // Q fragments for this wave's 16 rows: resident all kernel
  v16bf qf[4];
#pragma unroll
  for (int ks = 0; ks < 4; ++ks)
    qf[ks] = frag_ld(qh + ((size_t)qrow << 7) + ks * 32, half);

  float m_r[8], l_r[8];
#pragma unroll
  for (int e = 0; e < 8; ++e) { m_r[e] = -3.0e38f; l_r[e] = 0.0f; }
  v8f acc[8] = {};

  for (int kb = 0; kb < S_LEN / 64; ++kb) {
    __syncthreads();
    // stage K via async-to-LDS; V transposed via registers
#pragma unroll
    for (int i = 0; i < 4; ++i) {
      int idx8 = i * 256 + t;             // 1024 x 16B = 8192 elems
      int row = idx8 >> 4;                // key 0..63
      int colb = (idx8 & 15) * 8;         // dh base
      const u16* ks_src = kh + (((size_t)(kb * 64 + row)) << 7) + colb;
      async_copy16(ks_src, &kt[row][colb]);
      uint4 pv4 = *reinterpret_cast<const uint4*>(
          vh + (((size_t)(kb * 64 + row)) << 7) + colb);
      const u16* pp = reinterpret_cast<const u16*>(&pv4);
#pragma unroll
      for (int j = 0; j < 8; ++j) vt[colb + j][row] = pp[j];
    }
    async_wait0();
    __syncthreads();

    // scores S = Q * K^T  (16 x 64), 16 WMMAs
    v8f sc[4];
#pragma unroll
    for (int nt = 0; nt < 4; ++nt) {
      v8f s = {};
#pragma unroll
      for (int ks = 0; ks < 4; ++ks) {
        v16bf bf = frag_ld(&kt[nt * 16 + lo][ks * 32], half);
        s = wmma_bf16(qf[ks], bf, s);
      }
      sc[nt] = s;
    }

    // online softmax: row max across tiles + 16-lane half reduction
    float bm[8];
#pragma unroll
    for (int e = 0; e < 8; ++e)
      bm[e] = fmaxf(fmaxf(sc[0][e], sc[1][e]), fmaxf(sc[2][e], sc[3][e]));
#pragma unroll
    for (int msk = 1; msk <= 8; msk <<= 1)
#pragma unroll
      for (int e = 0; e < 8; ++e)
        bm[e] = fmaxf(bm[e], __shfl_xor(bm[e], msk, 32));

    float mnew[8], scl[8];
#pragma unroll
    for (int e = 0; e < 8; ++e) {
      mnew[e] = fmaxf(m_r[e], bm[e]);
      scl[e] = __expf(m_r[e] - mnew[e]);
      m_r[e] = mnew[e];
    }
    float bs[8] = {};
#pragma unroll
    for (int nt = 0; nt < 4; ++nt)
#pragma unroll
      for (int e = 0; e < 8; ++e) {
        float p = __expf(sc[nt][e] - mnew[e]);
        sc[nt][e] = p;
        bs[e] += p;
      }
#pragma unroll
    for (int msk = 1; msk <= 8; msk <<= 1)
#pragma unroll
      for (int e = 0; e < 8; ++e) bs[e] += __shfl_xor(bs[e], msk, 32);
#pragma unroll
    for (int e = 0; e < 8; ++e) l_r[e] = l_r[e] * scl[e] + bs[e];
#pragma unroll
    for (int n = 0; n < 8; ++n)
#pragma unroll
      for (int e = 0; e < 8; ++e) acc[n][e] *= scl[e];

    // C-layout -> A-layout transpose of P through per-wave LDS (bf16)
#pragma unroll
    for (int nt = 0; nt < 4; ++nt)
#pragma unroll
      for (int e = 0; e < 8; ++e)
        pl[w][half * 8 + e][nt * 16 + lo] = f32_to_bf16(sc[nt][e]);
    asm volatile("s_wait_dscnt 0" ::: "memory");  // same-wave DS RAW fence

    // O += P * V  (16 x 128), 16 WMMAs
    v16bf pf[2];
#pragma unroll
    for (int ks2 = 0; ks2 < 2; ++ks2)
      pf[ks2] = frag_ld(&pl[w][lo][ks2 * 32], half);
#pragma unroll
    for (int n = 0; n < 8; ++n)
#pragma unroll
      for (int ks2 = 0; ks2 < 2; ++ks2) {
        v16bf vb = frag_ld(&vt[n * 16 + lo][ks2 * 32], half);
        acc[n] = wmma_bf16(pf[ks2], vb, acc[n]);
      }
  }

  // normalize and write concatenated heads (bf16) for output projection
#pragma unroll
  for (int n = 0; n < 8; ++n)
#pragma unroll
    for (int e = 0; e < 8; ++e) {
      int grow = qblk * 128 + w * 16 + half * 8 + e;
      int gcol = h * DH_DIM + n * 16 + lo;
      cat16[(size_t)grow * D_DIM + gcol] = f32_to_bf16(acc[n][e] / l_r[e]);
    }
}

// ---------------------------------------------------------------- out proj
__global__ __launch_bounds__(256) void out_proj_kernel(
    const u16* __restrict__ cat16, const u16* __restrict__ wp16,
    const float* __restrict__ bp, float* __restrict__ out) {
  __shared__ u16 a_lds[256][40];
  __shared__ u16 bt_lds[256][40];
  const int m0 = blockIdx.x * 128, n0 = blockIdx.y * 128;

  v8f acc[4][2] = {};
  gemm_accum(cat16, D_DIM, wp16, D_DIM, m0, n0, D_DIM, a_lds, bt_lds, acc);

  const int t = threadIdx.x, w = t >> 5, lane = t & 31, lo = lane & 15, half = lane >> 4;
  const int wm = w >> 2, wn = w & 3;
#pragma unroll
  for (int mi = 0; mi < 4; ++mi)
#pragma unroll
    for (int ni = 0; ni < 2; ++ni)
#pragma unroll
      for (int r = 0; r < 8; ++r) {
        int grow = m0 + wm * 64 + mi * 16 + half * 8 + r;
        int gcol = n0 + wn * 32 + ni * 16 + lo;
        out[(size_t)grow * D_DIM + gcol] = acc[mi][ni][r] + bp[gcol];
      }
}

// ---------------------------------------------------------------- launch
extern "C" void kernel_launch(void* const* d_in, const int* in_sizes, int n_in,
                              void* d_out, int out_size, void* d_ws, size_t ws_size,
                              hipStream_t stream) {
  const float* z  = (const float*)d_in[0];
  const float* Wq = (const float*)d_in[1];
  const float* bq = (const float*)d_in[2];
  const float* Wk = (const float*)d_in[3];
  const float* bk = (const float*)d_in[4];
  const float* Wv = (const float*)d_in[5];
  const float* bv = (const float*)d_in[6];
  const float* Wp = (const float*)d_in[7];
  const float* bp = (const float*)d_in[8];

  char* ws = (char*)d_ws;
  const size_t MB = 1u << 20;
  u16* z16  = (u16*)(ws + 0 * MB);   // 8 MB
  u16* wq16 = (u16*)(ws + 8 * MB);   // 2 MB
  u16* wk16 = (u16*)(ws + 10 * MB);  // 2 MB
  u16* wv16 = (u16*)(ws + 12 * MB);  // 2 MB
  u16* wp16 = (u16*)(ws + 14 * MB);  // 2 MB
  u16* q16  = (u16*)(ws + 16 * MB);  // 8 MB
  u16* k16  = (u16*)(ws + 24 * MB);  // 8 MB
  u16* v16  = (u16*)(ws + 32 * MB);  // 8 MB
  u16* c16  = (u16*)(ws + 40 * MB);  // 8 MB

  const int nz = S_LEN * D_DIM;              // 4M
  const int nw = H_NUM * D_DIM * DH_DIM;     // 1M
  const int np = D_DIM * D_DIM;              // 1M
  cvt_kernel<<<dim3((nz + 255) / 256), dim3(256), 0, stream>>>(z, z16, nz);
  cvt_kernel<<<dim3((nw + 255) / 256), dim3(256), 0, stream>>>(Wq, wq16, nw);
  cvt_kernel<<<dim3((nw + 255) / 256), dim3(256), 0, stream>>>(Wk, wk16, nw);
  cvt_kernel<<<dim3((nw + 255) / 256), dim3(256), 0, stream>>>(Wv, wv16, nw);
  cvt_kernel<<<dim3((np + 255) / 256), dim3(256), 0, stream>>>(Wp, wp16, np);

  qkv_proj_kernel<<<dim3(S_LEN / 128, H_NUM, 3), dim3(256), 0, stream>>>(
      z16, wq16, wk16, wv16, bq, bk, bv, q16, k16, v16);

  attn_kernel<<<dim3(S_LEN / 128, H_NUM), dim3(256), 0, stream>>>(
      q16, k16, v16, c16);

  out_proj_kernel<<<dim3(S_LEN / 128, D_DIM / 128), dim3(256), 0, stream>>>(
      c16, wp16, bp, (float*)d_out);
}
